// Receiver_56169582297449
// MI455X (gfx1250) — compile-verified, matrix operand order checked by reading, not run on hardware
//
#include <hip/hip_runtime.h>
#include <hip/hip_bf16.h>
#include <math.h>

// ---------------------------------------------------------------------------
// Model dims (fixed by reference)
// ---------------------------------------------------------------------------
#define BV 64      // vocab
#define EE 512     // embed
#define HH 1024    // hidden
#define G4 4096    // 4*H
#define BB 512     // batch
#define ML 30      // msg len / max len
#define NFNV 128   // NF*NV

#define KC   128           // K-chunk staged in LDS
#define BROW 136           // 128 + 8 f16 pad: row stride 272B kills bank conflicts

typedef __attribute__((ext_vector_type(16))) _Float16 v16h;
typedef __attribute__((ext_vector_type(8)))  _Float16 v8h;
typedef __attribute__((ext_vector_type(8)))  float    v8f;

union V16 { v16h v; v8h h[2]; };

// ---------------------------------------------------------------------------
// WMMA GEMM:  C[M,N] = A[M,K](f16) * W[N,K](f16)^T  (+bias[n])
// Block = 8 waves sharing one 64-col weight strip (tn); waves take tm0..tm0+7.
// Weight strip staged into LDS in KC-chunks via async global->LDS (ASYNCcnt),
// double-buffered.  Requires M%128==0, N%64==0, K%KC==0 (true for this model).
// ---------------------------------------------------------------------------
__device__ __forceinline__ void stage_B_async(const _Float16* __restrict__ W,
                                              int n0, int K, int kc,
                                              _Float16* smB /* [64][BROW] */) {
  int tid = threadIdx.x;
  // 64 rows x 128 f16 = 1024 x 16-byte segments; 4 async ops per wave
  #pragma unroll
  for (int it = 0; it < 4; ++it) {
    int s   = tid + it * 256;
    int row = s >> 4;
    int c16 = s & 15;
    const _Float16* gp = W + (size_t)(n0 + row) * K + (size_t)kc * KC + c16 * 8;
    unsigned int la = (unsigned int)(unsigned long long)(&smB[row * BROW + c16 * 8]);
    asm volatile("global_load_async_to_lds_b128 %0, %1, off"
                 :: "v"(la), "v"(gp) : "memory");
  }
}

__global__ __launch_bounds__(256) void gemm_f16_wmma(
    const _Float16* __restrict__ A, const _Float16* __restrict__ W,
    const float* __restrict__ bias, float* __restrict__ C,
    int M, int N, int K)
{
  __shared__ __align__(16) _Float16 smB[2][64 * BROW];

  int tmBlocks = M >> 7;                    // M/128
  int tn  = blockIdx.x / tmBlocks;          // 64-col strip, shared by block
  int tmg = blockIdx.x % tmBlocks;
  int wave = threadIdx.x >> 5;
  int lane = threadIdx.x & 31;
  int tm   = tmg * 8 + wave;                // 16-row tile per wave

  int r     = lane & 15;
  int khalf = (lane >> 4) << 4;             // 0 or 16
  int n0    = tn * 64;

  const _Float16* Ap = A + (size_t)(tm * 16 + r) * K + khalf;

  v8f acc0 = {}, acc1 = {}, acc2 = {}, acc3 = {};

  int nk = K / KC;
  stage_B_async(W, n0, K, 0, smB[0]);

  for (int kc = 0; kc < nk; ++kc) {
    int buf = kc & 1;
    if (kc + 1 < nk) {
      stage_B_async(W, n0, K, kc + 1, smB[buf ^ 1]);
      asm volatile("s_wait_asynccnt 0x4" ::: "memory");  // current chunk landed
    } else {
      asm volatile("s_wait_asynccnt 0x0" ::: "memory");
    }
    __syncthreads();

    const _Float16* Apc = Ap + (size_t)kc * KC;
    #pragma unroll
    for (int k0 = 0; k0 < KC; k0 += 32) {
      v16h a = *(const v16h*)(Apc + k0);
      const _Float16* bp0 = &smB[buf][(0 * 16 + r) * BROW + k0 + khalf];
      const _Float16* bp1 = &smB[buf][(1 * 16 + r) * BROW + k0 + khalf];
      const _Float16* bp2 = &smB[buf][(2 * 16 + r) * BROW + k0 + khalf];
      const _Float16* bp3 = &smB[buf][(3 * 16 + r) * BROW + k0 + khalf];
      V16 b0, b1, b2, b3;
      b0.h[0] = *(const v8h*)bp0;       b0.h[1] = *(const v8h*)(bp0 + 8);
      b1.h[0] = *(const v8h*)bp1;       b1.h[1] = *(const v8h*)(bp1 + 8);
      b2.h[0] = *(const v8h*)bp2;       b2.h[1] = *(const v8h*)(bp2 + 8);
      b3.h[0] = *(const v8h*)bp3;       b3.h[1] = *(const v8h*)(bp3 + 8);
      acc0 = __builtin_amdgcn_wmma_f32_16x16x32_f16(false, a, false, b0.v, (short)0, acc0, false, false);
      acc1 = __builtin_amdgcn_wmma_f32_16x16x32_f16(false, a, false, b1.v, (short)0, acc1, false, false);
      acc2 = __builtin_amdgcn_wmma_f32_16x16x32_f16(false, a, false, b2.v, (short)0, acc2, false, false);
      acc3 = __builtin_amdgcn_wmma_f32_16x16x32_f16(false, a, false, b3.v, (short)0, acc3, false, false);
    }
    __syncthreads();
  }

  // C layout: VGPR rr -> M = tm*16 + (lane<16 ? rr : 8+rr), N = tn*64 + jj*16 + (lane&15)
  int rb = tm * 16 + (lane >> 4) * 8;
  #pragma unroll
  for (int jj = 0; jj < 4; ++jj) {
    int col = tn * 64 + jj * 16 + (lane & 15);
    float bv = bias ? bias[col] : 0.0f;
    v8f acc = (jj == 0) ? acc0 : (jj == 1) ? acc1 : (jj == 2) ? acc2 : acc3;
    #pragma unroll
    for (int rr = 0; rr < 8; ++rr)
      C[(size_t)(rb + rr) * N + col] = acc[rr] + bv;
  }
}

// ---------------------------------------------------------------------------
// Utility kernels
// ---------------------------------------------------------------------------
__global__ void cvt_f32_f16(const float* __restrict__ s, _Float16* __restrict__ d, size_t n) {
  size_t i = blockIdx.x * (size_t)blockDim.x + threadIdx.x;
  size_t st = gridDim.x * (size_t)blockDim.x;
  for (; i < n; i += st) d[i] = (_Float16)s[i];
}
__global__ void fill_f32(float* p, size_t n) {
  size_t i = blockIdx.x * (size_t)blockDim.x + threadIdx.x;
  size_t st = gridDim.x * (size_t)blockDim.x;
  for (; i < n; i += st) p[i] = 0.0f;
}
__global__ void fill_f16(_Float16* p, size_t n) {
  size_t i = blockIdx.x * (size_t)blockDim.x + threadIdx.x;
  size_t st = gridDim.x * (size_t)blockDim.x;
  for (; i < n; i += st) p[i] = (_Float16)0.0f;
}

// token embedding -> f16 input buffer [B, 512]
__global__ __launch_bounds__(256) void embed_gather(
    const int* __restrict__ toks, int stride, int t,
    const float* __restrict__ emb, _Float16* __restrict__ x16)
{
  int b = blockIdx.x;
  int tok = toks[(size_t)b * stride + t];
  const float* src = emb + (size_t)tok * EE;
  _Float16* dst = x16 + (size_t)b * EE;
  for (int e = threadIdx.x; e < EE; e += 256) dst[e] = (_Float16)src[e];
}

__global__ __launch_bounds__(256) void sos_bcast(const float* __restrict__ sos,
                                                 _Float16* __restrict__ x16) {
  int b = blockIdx.x;
  for (int e = threadIdx.x; e < EE; e += 256) x16[(size_t)b * EE + e] = (_Float16)sos[e];
}

__global__ __launch_bounds__(256) void select_enc(const float* __restrict__ h,
                                                  const int* __restrict__ mlen, int t,
                                                  float* __restrict__ enc) {
  int b = blockIdx.x;
  if (mlen[b] - 1 != t) return;
  for (int e = threadIdx.x; e < HH; e += 256)
    enc[(size_t)b * HH + e] = h[(size_t)b * HH + e];
}

__global__ void zero_tail(float* seq, float* lg, float* ent) {
  int b = blockIdx.x * blockDim.x + threadIdx.x;
  if (b < BB) {
    seq[(size_t)b * (ML + 1) + ML] = 0.0f;
    lg [(size_t)b * (ML + 1) + ML] = 0.0f;
    ent[(size_t)b * (ML + 1) + ML] = 0.0f;
  }
}

// ---------------------------------------------------------------------------
// LayerNorm-LSTM cell: gates = LN(ig) + LN(hg); cy = LN(f*c + i*g); h = o*tanh(cy)
// one block (256 threads) per batch row
// ---------------------------------------------------------------------------
__device__ __forceinline__ float blockReduceSum(float v, float* sm) {
  int tid = threadIdx.x;
  sm[tid] = v; __syncthreads();
  for (int s = 128; s > 0; s >>= 1) {
    if (tid < s) sm[tid] += sm[tid + s];
    __syncthreads();
  }
  float r = sm[0]; __syncthreads();
  return r;
}

__global__ __launch_bounds__(256) void lstm_cell(
    const float* __restrict__ ig, const float* __restrict__ hg,
    const float* __restrict__ lni_g, const float* __restrict__ lni_b,
    const float* __restrict__ lnh_g, const float* __restrict__ lnh_b,
    const float* __restrict__ lnc_g, const float* __restrict__ lnc_b,
    float* __restrict__ h32, float* __restrict__ c32, _Float16* __restrict__ h16)
{
  __shared__ float sm[256];
  int b = blockIdx.x, tid = threadIdx.x;
  const float* igr = ig + (size_t)b * G4;
  const float* hgr = hg + (size_t)b * G4;

  float si = 0, sqi = 0, sh = 0, sqh = 0;
  for (int k = tid; k < G4; k += 256) {
    float a = igr[k]; si += a; sqi += a * a;
    float c = hgr[k]; sh += c; sqh += c * c;
  }
  si  = blockReduceSum(si,  sm);
  sqi = blockReduceSum(sqi, sm);
  sh  = blockReduceSum(sh,  sm);
  sqh = blockReduceSum(sqh, sm);
  float mi = si / (float)G4, mh = sh / (float)G4;
  float rsi = rsqrtf(sqi / (float)G4 - mi * mi + 1e-5f);
  float rsh = rsqrtf(sqh / (float)G4 - mh * mh + 1e-5f);

  float cp[4], oo[4];
  float sc = 0, sqc = 0;
  #pragma unroll
  for (int u = 0; u < 4; ++u) {
    int j = tid + u * 256;
    auto zz = [&](int k) {
      return (igr[k] - mi) * rsi * lni_g[k] + lni_b[k]
           + (hgr[k] - mh) * rsh * lnh_g[k] + lnh_b[k];
    };
    float iv = 1.0f / (1.0f + expf(-zz(j)));
    float fv = 1.0f / (1.0f + expf(-zz(j + HH)));
    float gv = tanhf(zz(j + 2 * HH));
    float ov = 1.0f / (1.0f + expf(-zz(j + 3 * HH)));
    float c0 = c32[(size_t)b * HH + j];
    float cpv = fv * c0 + iv * gv;
    cp[u] = cpv; oo[u] = ov;
    sc += cpv; sqc += cpv * cpv;
  }
  sc  = blockReduceSum(sc,  sm);
  sqc = blockReduceSum(sqc, sm);
  float mc = sc / (float)HH;
  float rsc = rsqrtf(sqc / (float)HH - mc * mc + 1e-5f);

  #pragma unroll
  for (int u = 0; u < 4; ++u) {
    int j = tid + u * 256;
    float cy = (cp[u] - mc) * rsc * lnc_g[j] + lnc_b[j];
    float hv = oo[u] * tanhf(cy);
    c32[(size_t)b * HH + j] = cy;
    h32[(size_t)b * HH + j] = hv;
    h16[(size_t)b * HH + j] = (_Float16)hv;
  }
}

// ---------------------------------------------------------------------------
// Decoder head: log_softmax over V=64, argmax, entropy, logprob, next embedding
// one block (64 threads) per batch row
// ---------------------------------------------------------------------------
__global__ __launch_bounds__(64) void dec_head(
    const float* __restrict__ head, const float* __restrict__ emb_prod,
    _Float16* __restrict__ x16, float* __restrict__ seq, float* __restrict__ lg,
    float* __restrict__ ent, float* __restrict__ asl, int t)
{
  __shared__ float sv[64];
  __shared__ int   si[64];
  __shared__ int   sx;
  __shared__ float smx, ssum;
  int b = blockIdx.x, j = threadIdx.x;
  float z = head[(size_t)b * BV + j];

  sv[j] = z; si[j] = j; __syncthreads();
  for (int s = 32; s > 0; s >>= 1) {
    if (j < s && sv[j + s] > sv[j]) { sv[j] = sv[j + s]; si[j] = si[j + s]; }
    __syncthreads();
  }
  if (j == 0) { smx = sv[0]; sx = si[0]; }
  __syncthreads();
  float mx = smx; int amax = sx;

  sv[j] = expf(z - mx); __syncthreads();
  for (int s = 32; s > 0; s >>= 1) { if (j < s) sv[j] += sv[j + s]; __syncthreads(); }
  if (j == 0) ssum = sv[0];
  __syncthreads();
  float lZ = logf(ssum);
  float sl = z - mx - lZ;
  asl[((size_t)b * ML + t) * BV + j] = sl;

  sv[j] = expf(sl) * sl; __syncthreads();
  for (int s = 32; s > 0; s >>= 1) { if (j < s) sv[j] += sv[j + s]; __syncthreads(); }
  if (j == 0) {
    ent[(size_t)b * (ML + 1) + t] = -sv[0];
    seq[(size_t)b * (ML + 1) + t] = (float)amax;
    lg [(size_t)b * (ML + 1) + t] = head[(size_t)b * BV + amax] - mx - lZ;
  }
  // next input = emb_prod[amax], converted to f16
  const float* src = emb_prod + (size_t)amax * EE;
  _Float16* dst = x16 + (size_t)b * EE;
  for (int e2 = j; e2 < EE; e2 += 64) dst[e2] = (_Float16)src[e2];
}

// ---------------------------------------------------------------------------
// Host orchestration
// ---------------------------------------------------------------------------
struct CellP { const float *wih, *whh, *lni_g, *lni_b, *lnh_g, *lnh_b, *lnc_g, *lnc_b; };

extern "C" void kernel_launch(void* const* d_in, const int* in_sizes, int n_in,
                              void* d_out, int out_size, void* d_ws, size_t ws_size,
                              hipStream_t stream) {
  // inputs in setup_inputs() order; nested dicts flattened in declaration order
  const int*   messages = (const int*)d_in[0];
  const int*   mlen     = (const int*)d_in[1];
  const float* emb_perc = (const float*)d_in[2];
  const float* emb_prod = (const float*)d_in[3];
  const float* sos      = (const float*)d_in[4];
  auto getCell = [&](int base) {
    CellP c;
    c.wih   = (const float*)d_in[base + 0];
    c.whh   = (const float*)d_in[base + 1];
    c.lni_g = (const float*)d_in[base + 2];
    c.lni_b = (const float*)d_in[base + 3];
    c.lnh_g = (const float*)d_in[base + 4];
    c.lnh_b = (const float*)d_in[base + 5];
    c.lnc_g = (const float*)d_in[base + 6];
    c.lnc_b = (const float*)d_in[base + 7];
    return c;
  };
  CellP p0 = getCell(5),  p1 = getCell(13);   // perceiver (encoder) layers
  CellP q0 = getCell(21), q1 = getCell(29);   // producer (decoder) layers
  const float* w_out = (const float*)d_in[37];
  const float* b_out = (const float*)d_in[38];
  const float* w_msg = (const float*)d_in[39];
  const float* b_msg = (const float*)d_in[40];
  const float* w_ho  = (const float*)d_in[41];
  const float* b_ho  = (const float*)d_in[42];

  // output regions (f32, concatenated in return order)
  float* out      = (float*)d_out;
  float* o_output = out;                       // [512,128]
  float* o_seq    = out + 65536;               // [512,31]
  float* o_logits = out + 81408;               // [512,31]
  float* o_ent    = out + 97280;               // [512,31]
  float* o_asl    = out + 113152;              // [512,30,64]
  float* o_enc    = out + 1096192;             // [512,1024]

  // workspace carve
  char* wp = (char*)d_ws;
  auto a16 = [&](size_t n) { _Float16* r = (_Float16*)wp; wp += ((n * 2 + 255) / 256) * 256; return r; };
  auto a32 = [&](size_t n) { float*     r = (float*)wp;   wp += ((n * 4 + 255) / 256) * 256; return r; };

  _Float16 *wih_p0 = a16((size_t)G4 * EE), *whh_p0 = a16((size_t)G4 * HH);
  _Float16 *wih_p1 = a16((size_t)G4 * HH), *whh_p1 = a16((size_t)G4 * HH);
  _Float16 *wih_q0 = a16((size_t)G4 * EE), *whh_q0 = a16((size_t)G4 * HH);
  _Float16 *wih_q1 = a16((size_t)G4 * HH), *whh_q1 = a16((size_t)G4 * HH);
  _Float16 *wout16 = a16((size_t)NFNV * HH);
  _Float16 *wmsg16 = a16((size_t)HH * HH);
  _Float16 *who16  = a16((size_t)BV * HH);
  _Float16 *x16    = a16((size_t)BB * EE);
  _Float16 *h16_0  = a16((size_t)BB * HH);
  _Float16 *h16_1  = a16((size_t)BB * HH);
  _Float16 *enc16  = a16((size_t)BB * HH);
  float *ig32  = a32((size_t)BB * G4);
  float *hg32  = a32((size_t)BB * G4);
  float *h32_0 = a32((size_t)BB * HH), *h32_1 = a32((size_t)BB * HH);
  float *c32_0 = a32((size_t)BB * HH), *c32_1 = a32((size_t)BB * HH);
  float *head32 = a32((size_t)BB * BV);

  auto cvt = [&](const float* s, _Float16* d, size_t n) {
    int blocks = (int)((n + 255) / 256); if (blocks > 2048) blocks = 2048;
    cvt_f32_f16<<<blocks, 256, 0, stream>>>(s, d, n);
  };
  auto z32 = [&](float* p, size_t n) {
    int blocks = (int)((n + 255) / 256); if (blocks > 2048) blocks = 2048;
    fill_f32<<<blocks, 256, 0, stream>>>(p, n);
  };
  auto z16 = [&](_Float16* p, size_t n) {
    int blocks = (int)((n + 255) / 256); if (blocks > 2048) blocks = 2048;
    fill_f16<<<blocks, 256, 0, stream>>>(p, n);
  };
  auto gemm = [&](const _Float16* A, const _Float16* W, const float* bias,
                  float* C, int M, int N, int K) {
    int blocks = (N / 64) * (M / 128);
    gemm_f16_wmma<<<blocks, 256, 0, stream>>>(A, W, bias, C, M, N, K);
  };
  auto cell = [&](const CellP& P, float* h32, float* c32, _Float16* h16) {
    lstm_cell<<<BB, 256, 0, stream>>>(ig32, hg32, P.lni_g, P.lni_b, P.lnh_g, P.lnh_b,
                                      P.lnc_g, P.lnc_b, h32, c32, h16);
  };

  // ---- weight conversion (every call; deterministic, no caching allowed) ----
  cvt(p0.wih, wih_p0, (size_t)G4 * EE);  cvt(p0.whh, whh_p0, (size_t)G4 * HH);
  cvt(p1.wih, wih_p1, (size_t)G4 * HH);  cvt(p1.whh, whh_p1, (size_t)G4 * HH);
  cvt(q0.wih, wih_q0, (size_t)G4 * EE);  cvt(q0.whh, whh_q0, (size_t)G4 * HH);
  cvt(q1.wih, wih_q1, (size_t)G4 * HH);  cvt(q1.whh, whh_q1, (size_t)G4 * HH);
  cvt(w_out, wout16, (size_t)NFNV * HH);
  cvt(w_msg, wmsg16, (size_t)HH * HH);
  cvt(w_ho,  who16,  (size_t)BV * HH);

  // ---- encoder init ----
  z32(h32_0, (size_t)BB * HH); z32(h32_1, (size_t)BB * HH);
  z32(c32_0, (size_t)BB * HH); z32(c32_1, (size_t)BB * HH);
  z16(h16_0, (size_t)BB * HH); z16(h16_1, (size_t)BB * HH);
  zero_tail<<<2, 256, 0, stream>>>(o_seq, o_logits, o_ent);

  // ---- encoder: 30 steps of 2-layer LN-LSTM ----
  for (int t = 0; t < ML; ++t) {
    embed_gather<<<BB, 256, 0, stream>>>(messages, ML, t, emb_perc, x16);
    // layer 0
    gemm(x16,   wih_p0, nullptr, ig32, BB, G4, EE);
    gemm(h16_0, whh_p0, nullptr, hg32, BB, G4, HH);
    cell(p0, h32_0, c32_0, h16_0);
    // layer 1 (input = layer-0 h, already f16 in h16_0)
    gemm(h16_0, wih_p1, nullptr, ig32, BB, G4, HH);
    gemm(h16_1, whh_p1, nullptr, hg32, BB, G4, HH);
    cell(p1, h32_1, c32_1, h16_1);
    // capture encoded state at t == mlen-1
    select_enc<<<BB, 256, 0, stream>>>(h32_1, mlen, t, o_enc);
  }

  // ---- heads on encoded ----
  cvt(o_enc, enc16, (size_t)BB * HH);
  gemm(enc16, wout16, b_out, o_output, BB, NFNV, HH);   // output = enc @ w_out^T + b_out
  gemm(enc16, wmsg16, b_msg, h32_0,    BB, HH,   HH);   // hdec[0] = enc @ w_msg^T + b_msg

  // ---- decoder init ----
  cvt(h32_0, h16_0, (size_t)BB * HH);
  z32(h32_1, (size_t)BB * HH);
  z32(c32_0, (size_t)BB * HH); z32(c32_1, (size_t)BB * HH);
  z16(h16_1, (size_t)BB * HH);
  sos_bcast<<<BB, 256, 0, stream>>>(sos, x16);

  // ---- decoder: 30 greedy steps ----
  for (int t = 0; t < ML; ++t) {
    // layer 0
    gemm(x16,   wih_q0, nullptr, ig32, BB, G4, EE);
    gemm(h16_0, whh_q0, nullptr, hg32, BB, G4, HH);
    cell(q0, h32_0, c32_0, h16_0);
    // layer 1
    gemm(h16_0, wih_q1, nullptr, ig32, BB, G4, HH);
    gemm(h16_1, whh_q1, nullptr, hg32, BB, G4, HH);
    cell(q1, h32_1, c32_1, h16_1);
    // vocab head + log-softmax/argmax/entropy + next embedding
    gemm(h16_1, who16, b_ho, head32, BB, BV, HH);
    dec_head<<<BB, 64, 0, stream>>>(head32, emb_prod, x16,
                                    o_seq, o_logits, o_ent, o_asl, t);
  }
}
